// GAT_20194936225911
// MI455X (gfx1250) — matrix operand order repaired and checked by reading, not compiled, not run on hardware
//
#include <hip/hip_runtime.h>

#define NNODES 100000
#define NEDGES 1600000
#define TLEN   35
#define CTC    16
#define HCF    128
#define NHEAD  4
#define BGRAPH 64
#define LEAKY  0.2f

typedef __attribute__((ext_vector_type(16))) _Float16 v16h;
typedef __attribute__((ext_vector_type(8)))  float    v8f;

__device__ __forceinline__ float lrelu(float v) { return v > 0.f ? v : LEAKY * v; }

__device__ __forceinline__ void atomicMaxF(float* addr, float val) {
    int* ai = (int*)addr;
    int old = __float_as_int(*addr);
    while (__int_as_float(old) < val) {
        int assumed = old;
        old = atomicCAS(ai, assumed, __float_as_int(val));
        if (old == assumed) break;
    }
}

// ---------------------------------------------------------------- utilities
__global__ void k_fill(float* __restrict__ p, float v, int n) {
    int i = blockIdx.x * blockDim.x + threadIdx.x;
    if (i < n) p[i] = v;
}

// transpose + f16 convert a [128,128] weight:  Wt[n*128+k] = W[k*128+n]
__global__ void k_convert_w(const float* __restrict__ W, _Float16* __restrict__ Wt) {
    int i = blockIdx.x * blockDim.x + threadIdx.x;
    if (i < HCF * HCF) {
        int k = i >> 7, n = i & 127;
        Wt[n * HCF + k] = (_Float16)W[k * HCF + n];
    }
}

// --------------------------------------------------------- temporal convs
__global__ __launch_bounds__(256) void k_temporal_conv(
    const float* __restrict__ x,
    const float* __restrict__ w1, const float* __restrict__ b1,
    const float* __restrict__ w2, const float* __restrict__ b2,
    _Float16* __restrict__ h16)
{
    __shared__ float s1[16][CTC][17];
    int nodeLocal = threadIdx.x >> 4;
    int j = threadIdx.x & 15;
    int n = blockIdx.x * 16 + nodeLocal;
    int nc = n < NNODES ? n : NNODES - 1;
    const float* xr = x + (size_t)nc * TLEN;

    // stage 1: j = conv1 channel; out length 17, stride 2, k=3
    {
        float wa = w1[j * 3 + 0], wb = w1[j * 3 + 1], wc = w1[j * 3 + 2], bb = b1[j];
        #pragma unroll
        for (int q = 0; q < 17; ++q) {
            float v = bb + wa * xr[2 * q] + wb * xr[2 * q + 1] + wc * xr[2 * q + 2];
            s1[nodeLocal][j][q] = fmaxf(v, 0.f);
        }
    }
    __syncthreads();
    // stage 2: j = conv2 out channel; out length 8; feature idx = co*8 + p
    if (n < NNODES) {
        #pragma unroll
        for (int p = 0; p < 8; ++p) {
            float acc = b2[j];
            #pragma unroll
            for (int ci = 0; ci < CTC; ++ci) {
                const float* wr = w2 + (j * CTC + ci) * 3;
                acc += wr[0] * s1[nodeLocal][ci][2 * p]
                     + wr[1] * s1[nodeLocal][ci][2 * p + 1]
                     + wr[2] * s1[nodeLocal][ci][2 * p + 2];
            }
            h16[(size_t)n * HCF + j * 8 + p] = (_Float16)fmaxf(acc, 0.f);
        }
    }
}

// -------------------------------------- dual-weight WMMA GEMM  Yl/Yr = Xh @ Wl/Wr
// Xh: [M,128] f16 row-major.  W*tH: transposed f16 weights, WtH[n*128+k] = W[k][n].
// One wave computes a 16-row strip x all 128 cols for BOTH weight matrices
// (A fragments shared), K = 128 consumed in 4 chunks of 32.
__global__ __launch_bounds__(128) void k_gemm128_wmma_dual(
    const _Float16* __restrict__ Xh,
    const _Float16* __restrict__ WltH, const _Float16* __restrict__ WrtH,
    float* __restrict__ Yl, float* __restrict__ Yr, int Mrows)
{
    int wave = blockIdx.x * (blockDim.x >> 5) + (threadIdx.x >> 5);
    int row0 = wave * 16;
    if (row0 >= Mrows) return;
    int lane = threadIdx.x & 31;
    int half = lane >> 4;      // which half-wave
    int m    = lane & 15;      // row within strip (A) / column within tile (B)

    const _Float16* xrow = Xh + (size_t)(row0 + m) * HCF;

    v8f accL[8], accR[8];
    #pragma unroll
    for (int nt = 0; nt < 8; ++nt) {
        v8f z = {0.f, 0.f, 0.f, 0.f, 0.f, 0.f, 0.f, 0.f};
        accL[nt] = z;
        accR[nt] = z;
    }

    #pragma unroll
    for (int kc = 0; kc < 4; ++kc) {
        // A fragment (16x32 f16): lanes 0-15: K = base+0..15
        //                         lanes 16-31: K = base+16..31
        v16h a;
        const int kbA = kc * 32 + half * 16;
        #pragma unroll
        for (int i = 0; i < 8; ++i) {
            a[i]     = xrow[kbA + i];
            a[8 + i] = xrow[kbA + 8 + i];
        }
        const int kbB = kc * 32 + half * 16;
        #pragma unroll
        for (int nt = 0; nt < 8; ++nt) {
            // B fragment (32x16): lane holds col n=m; K contiguous in transposed W
            const _Float16* wl = WltH + (size_t)(nt * 16 + m) * HCF + kbB;
            v16h bl;
            #pragma unroll
            for (int i = 0; i < 16; ++i) bl[i] = wl[i];
            accL[nt] = __builtin_amdgcn_wmma_f32_16x16x32_f16(
                false, a, false, bl, (short)0, accL[nt], false, false);

            const _Float16* wr = WrtH + (size_t)(nt * 16 + m) * HCF + kbB;
            v16h br;
            #pragma unroll
            for (int i = 0; i < 16; ++i) br[i] = wr[i];
            accR[nt] = __builtin_amdgcn_wmma_f32_16x16x32_f16(
                false, a, false, br, (short)0, accR[nt], false, false);
        }
    }

    // C/D layout: VGPR r -> (M = r + half*8, N = m)
    #pragma unroll
    for (int nt = 0; nt < 8; ++nt) {
        #pragma unroll
        for (int r = 0; r < 8; ++r) {
            size_t off = (size_t)(row0 + r + half * 8) * HCF + nt * 16 + m;
            Yl[off] = accL[nt][r];
            Yr[off] = accR[nt][r];
        }
    }
}

// ------------------------------------------------------------ edge phase
__global__ void k_degree(const int* __restrict__ ei, const float* __restrict__ ea,
                         float* __restrict__ deg, float* __restrict__ easum) {
    int e = blockIdx.x * blockDim.x + threadIdx.x;
    if (e < NEDGES) {
        int d = ei[NEDGES + e];
        atomicAdd(&deg[d], 1.0f);
        atomicAdd(&easum[d], ea[e]);
    }
}

__global__ void k_loopattr(const float* __restrict__ deg, const float* __restrict__ easum,
                           float* __restrict__ la) {
    int n = blockIdx.x * blockDim.x + threadIdx.x;
    if (n < NNODES) la[n] = easum[n] / fmaxf(deg[n], 1.0f);
}

// pass 1: per-edge 4-head logits + segment max.  one wave per edge, 4 feats/lane.
__global__ __launch_bounds__(256) void k_edge_logits(
    const int* __restrict__ ei, const float* __restrict__ ea, const float* __restrict__ la,
    const float* __restrict__ msg, const float* __restrict__ tgt,
    const float* __restrict__ We, const float* __restrict__ att,
    float* __restrict__ logitbuf, float* __restrict__ nodemax)
{
    const int total = NEDGES + NNODES;
    int e = blockIdx.x * (blockDim.x >> 5) + (threadIdx.x >> 5);
    if (e >= total) return;
    int lane = threadIdx.x & 31;

    int s, d; float a;
    if (e < NEDGES) { s = ei[e]; d = ei[NEDGES + e]; a = ea[e]; }
    else            { s = d = e - NEDGES; a = la[s]; }

    const float4 mv = ((const float4*)(msg + (size_t)s * HCF))[lane];
    const float4 tv = ((const float4*)(tgt + (size_t)d * HCF))[lane];
    const float4 wv = ((const float4*)We)[lane];
    const float4 av = ((const float4*)att)[lane];

    float v0 = lrelu(mv.x + tv.x + a * wv.x);
    float v1 = lrelu(mv.y + tv.y + a * wv.y);
    float v2 = lrelu(mv.z + tv.z + a * wv.z);
    float v3 = lrelu(mv.w + tv.w + a * wv.w);
    float dotp = v0 * av.x + v1 * av.y + v2 * av.z + v3 * av.w;

    dotp += __shfl_xor(dotp, 1, 32);
    dotp += __shfl_xor(dotp, 2, 32);
    dotp += __shfl_xor(dotp, 4, 32);   // sum within 8-lane head group

    if ((lane & 7) == 0) {
        int h = lane >> 3;
        logitbuf[(size_t)e * NHEAD + h] = dotp;
        atomicMaxF(&nodemax[(size_t)d * NHEAD + h], dotp);
    }
}

// pass 2: p = exp(logit - max[dst]); segment sum.  one thread per (edge, head).
__global__ void k_edge_expsum(const int* __restrict__ ei,
                              float* __restrict__ logitbuf,
                              const float* __restrict__ nodemax,
                              float* __restrict__ nodesum)
{
    const int total = (NEDGES + NNODES) * NHEAD;
    int i = blockIdx.x * blockDim.x + threadIdx.x;
    if (i >= total) return;
    int e = i >> 2, h = i & 3;
    int d = (e < NEDGES) ? ei[NEDGES + e] : (e - NEDGES);
    float p = expf(logitbuf[i] - nodemax[(size_t)d * NHEAD + h]);
    logitbuf[i] = p;
    atomicAdd(&nodesum[(size_t)d * NHEAD + h], p);
}

// pass 3: out[dst] += msg[src] * alpha.  one wave per edge.
__global__ __launch_bounds__(256) void k_edge_aggregate(
    const int* __restrict__ ei,
    const float* __restrict__ logitbuf, const float* __restrict__ nodesum,
    const float* __restrict__ msg, float* __restrict__ outb)
{
    const int total = NEDGES + NNODES;
    int e = blockIdx.x * (blockDim.x >> 5) + (threadIdx.x >> 5);
    if (e >= total) return;
    int lane = threadIdx.x & 31;

    int s, d;
    if (e < NEDGES) { s = ei[e]; d = ei[NEDGES + e]; }
    else            { s = d = e - NEDGES; }

    int h = lane >> 3;
    float alpha = logitbuf[(size_t)e * NHEAD + h] / nodesum[(size_t)d * NHEAD + h];

    const float4 mv = ((const float4*)(msg + (size_t)s * HCF))[lane];
    float* ob = outb + (size_t)d * HCF + lane * 4;
    atomicAdd(ob + 0, mv.x * alpha);
    atomicAdd(ob + 1, mv.y * alpha);
    atomicAdd(ob + 2, mv.z * alpha);
    atomicAdd(ob + 3, mv.w * alpha);
}

__global__ void k_bias_relu(const float* __restrict__ acc, const float* __restrict__ bias,
                            float* __restrict__ h32, _Float16* __restrict__ h16)
{
    int i = blockIdx.x * blockDim.x + threadIdx.x;
    if (i < NNODES * HCF) {
        float v = fmaxf(acc[i] + bias[i & 127], 0.f);
        h32[i] = v;
        h16[i] = (_Float16)v;
    }
}

// ------------------------------------------------------------- pool + fc
__global__ void k_pool(const float* __restrict__ h, const int* __restrict__ batch,
                       float* __restrict__ g)
{
    int i = blockIdx.x * blockDim.x + threadIdx.x;
    if (i < NNODES * HCF) {
        int n = i >> 7, f = i & 127;
        atomicAdd(&g[(size_t)batch[n] * HCF + f], h[i]);
    }
}

__global__ void k_fc(const float* __restrict__ g, const float* __restrict__ fcw,
                     const float* __restrict__ fcb, float* __restrict__ out)
{
    int b = threadIdx.x;
    if (b < BGRAPH) {
        float acc = fcb[0];
        #pragma unroll
        for (int j = 0; j < HCF; ++j) acc += g[b * HCF + j] * fcw[j];
        out[b] = acc;
    }
}

// ---------------------------------------------------------------- driver
extern "C" void kernel_launch(void* const* d_in, const int* in_sizes, int n_in,
                              void* d_out, int out_size, void* d_ws, size_t ws_size,
                              hipStream_t stream) {
    (void)in_sizes; (void)n_in; (void)out_size; (void)ws_size;

    const float* x       = (const float*)d_in[0];
    const int*   ei      = (const int*)  d_in[1];   // [2,E] flat: src then dst
    const float* ea      = (const float*)d_in[2];
    const int*   batch   = (const int*)  d_in[3];
    const float* conv1_w = (const float*)d_in[4];
    const float* conv1_b = (const float*)d_in[5];
    const float* conv2_w = (const float*)d_in[6];
    const float* conv2_b = (const float*)d_in[7];
    const float* gW[2][5] = {
        {(const float*)d_in[8],  (const float*)d_in[9],  (const float*)d_in[10],
         (const float*)d_in[11], (const float*)d_in[12]},
        {(const float*)d_in[13], (const float*)d_in[14], (const float*)d_in[15],
         (const float*)d_in[16], (const float*)d_in[17]} };
    const float* fc_w = (const float*)d_in[18];
    const float* fc_b = (const float*)d_in[19];
    float* outp = (float*)d_out;

    // workspace carve-up (256B aligned)
    char* wp = (char*)d_ws;
    auto carve = [&](size_t bytes) -> void* {
        void* p = (void*)wp;
        wp += (bytes + 255) & ~(size_t)255;
        return p;
    };
    float*    h32      = (float*)   carve((size_t)NNODES * HCF * 4);
    _Float16* h16      = (_Float16*)carve((size_t)NNODES * HCF * 2);
    float*    msg      = (float*)   carve((size_t)NNODES * HCF * 4);
    float*    tgt      = (float*)   carve((size_t)NNODES * HCF * 4);
    float*    agg      = tgt;       // alias: tgt dead after k_edge_logits
    float*    deg      = (float*)   carve((size_t)NNODES * 4);
    float*    easum    = (float*)   carve((size_t)NNODES * 4);
    float*    la       = (float*)   carve((size_t)NNODES * 4);
    float*    logitbuf = (float*)   carve((size_t)(NEDGES + NNODES) * NHEAD * 4);
    float*    nodemax  = (float*)   carve((size_t)NNODES * NHEAD * 4);
    float*    nodesum  = (float*)   carve((size_t)NNODES * NHEAD * 4);
    _Float16* WH[4];
    for (int i = 0; i < 4; ++i)
        WH[i] = (_Float16*)carve((size_t)HCF * HCF * 2);   // g1_Wl, g1_Wr, g2_Wl, g2_Wr
    float*    gpool    = (float*)   carve((size_t)BGRAPH * HCF * 4);

    const int total  = NEDGES + NNODES;
    const int ewaveB = (total + 7) / 8;                 // 8 edge-waves per 256-thr block
    const int nfB    = (NNODES * HCF + 255) / 256;
    const int strips = (NNODES + 15) / 16;
    const int gemmB  = (strips + 3) / 4;                // 4 waves per 128-thr block

    // 1) temporal convs -> h16
    k_temporal_conv<<<(NNODES + 15) / 16, 256, 0, stream>>>(
        x, conv1_w, conv1_b, conv2_w, conv2_b, h16);

    // 2) degree / mean self-loop edge attr (shared by both GAT layers)
    k_fill<<<(NNODES + 255) / 256, 256, 0, stream>>>(deg, 0.f, NNODES);
    k_fill<<<(NNODES + 255) / 256, 256, 0, stream>>>(easum, 0.f, NNODES);
    k_degree<<<(NEDGES + 255) / 256, 256, 0, stream>>>(ei, ea, deg, easum);
    k_loopattr<<<(NNODES + 255) / 256, 256, 0, stream>>>(deg, easum, la);

    // 3) convert all four 128x128 weights once (transposed f16)
    for (int layer = 0; layer < 2; ++layer) {
        k_convert_w<<<(HCF * HCF + 255) / 256, 256, 0, stream>>>(gW[layer][0], WH[layer * 2 + 0]);
        k_convert_w<<<(HCF * HCF + 255) / 256, 256, 0, stream>>>(gW[layer][1], WH[layer * 2 + 1]);
    }

    // 4) two GATv2 layers
    for (int layer = 0; layer < 2; ++layer) {
        const float* We  = gW[layer][2];
        const float* att = gW[layer][3];
        const float* bia = gW[layer][4];

        // fused: msg = h@Wl, tgt = h@Wr (A fragments shared)
        k_gemm128_wmma_dual<<<gemmB, 128, 0, stream>>>(
            h16, WH[layer * 2 + 0], WH[layer * 2 + 1], msg, tgt, NNODES);

        k_fill<<<(NNODES * NHEAD + 255) / 256, 256, 0, stream>>>(nodemax, -3.0e38f, NNODES * NHEAD);
        k_fill<<<(NNODES * NHEAD + 255) / 256, 256, 0, stream>>>(nodesum, 0.f, NNODES * NHEAD);

        k_edge_logits<<<ewaveB, 256, 0, stream>>>(ei, ea, la, msg, tgt, We, att,
                                                  logitbuf, nodemax);
        // tgt no longer needed -> reuse as aggregation buffer
        k_fill<<<nfB, 256, 0, stream>>>(agg, 0.f, NNODES * HCF);
        k_edge_expsum<<<(total * NHEAD + 255) / 256, 256, 0, stream>>>(ei, logitbuf,
                                                                       nodemax, nodesum);
        k_edge_aggregate<<<ewaveB, 256, 0, stream>>>(ei, logitbuf, nodesum, msg, agg);
        k_bias_relu<<<nfB, 256, 0, stream>>>(agg, bia, h32, h16);
    }

    // 5) global add pool + fc
    k_fill<<<(BGRAPH * HCF + 255) / 256, 256, 0, stream>>>(gpool, 0.f, BGRAPH * HCF);
    k_pool<<<nfB, 256, 0, stream>>>(h32, batch, gpool);
    k_fc<<<1, 64, 0, stream>>>(gpool, fc_w, fc_b, outp);
}
